// StereoConv2d_65412351918200
// MI455X (gfx1250) — matrix operand order repaired and checked by reading, not compiled, not run on hardware
//
#include <hip/hip_runtime.h>
#include <hip/hip_bf16.h>

typedef __bf16 bf16_t;
typedef __attribute__((ext_vector_type(16))) __bf16 v16bf;
typedef __attribute__((ext_vector_type(8)))  float  v8f;

#define B_N   2
#define CIN_  64
#define HIN_  128
#define WIN_  256
#define COUT_ 64
#define KK_   9
#define HOUT_ 128
#define WOUT_ 256
#define HW_IN   (HIN_*WIN_)
#define HW_OUT  (HOUT_*WOUT_)
#define PIX_TILE 64

// One-time: weight[o][c][3][3] f32 -> wbf[k][cout][cin] bf16 (GEMM-B friendly layout)
__global__ __launch_bounds__(256)
void dcn_wcvt_kernel(const float* __restrict__ w, bf16_t* __restrict__ wbf) {
  int i = blockIdx.x * 256 + threadIdx.x;          // i = k*4096 + o*64 + c
  if (i >= KK_ * COUT_ * CIN_) return;
  int c = i & 63, o = (i >> 6) & 63, k = i >> 12;
  wbf[i] = (bf16_t)w[(o * CIN_ + c) * KK_ + k];
}

struct __align__(16) Param {
  float4 w;          // weights for cols (bx, bx+1) of rows (y0, y1)
  int o0, o1;        // element offsets of (y0,bx) and (y1,bx) within a channel
  int pad0, pad1;
};

__global__ __launch_bounds__(256)
void dcn_main_kernel(const float* __restrict__ x, const float* __restrict__ off,
                     const bf16_t* __restrict__ wbf, const float* __restrict__ bias,
                     float* __restrict__ out) {
  __shared__ bf16_t sA[2][PIX_TILE * 64];  // ping-pong sampled tiles (2x8KB)
  __shared__ bf16_t sW[2][64 * 64];        // ping-pong weight slices (2x8KB)
  __shared__ Param  sP[KK_][PIX_TILE];     // bilinear params          (18KB)

  const int t    = threadIdx.x;
  const int lane = t & 31;
  const int wave = t >> 5;
  const int hl   = lane & 15;
  const int hi   = lane >> 4;
  const int ps0  = wave & 1;            // pixel 16-subtiles: ps0 and ps0+2
  const int ps1  = (wave & 1) + 2;
  const int cs   = wave >> 1;           // cout 16-subtile (0..3)

  const int gp0 = blockIdx.x * PIX_TILE;     // 64 px: same batch, same row
  const int b   = gp0 >> 15;
  const float* xb = x + b * (CIN_ * HW_IN);

  // ---- setup: all 9 taps x 64 pixels of bilinear params, once per block ----
  for (int idx = t; idx < KK_ * PIX_TILE; idx += 256) {
    int k = idx >> 6, p = idx & 63;
    int gp = gp0 + p;
    int hw = gp & 32767, ho = hw >> 8, wo = hw & 255;
    const float* ob = off + ((long)b * (2 * KK_) + 2 * k) * HW_OUT + ho * WOUT_ + wo;
    float oy = ob[0];
    float ox = ob[HW_OUT];
    float py = (float)(ho - 1 + k / 3) + oy;
    float px = (float)(wo - 1 + k % 3) + ox;
    float y0f = floorf(py), x0f = floorf(px);
    float wy = py - y0f, wx = px - x0f;
    int y0 = (int)y0f, x0 = (int)x0f;
    int y1 = y0 + 1,   x1 = x0 + 1;
    float vy0 = (y0 >= 0 && y0 < HIN_) ? 1.f : 0.f;
    float vy1 = (y1 >= 0 && y1 < HIN_) ? 1.f : 0.f;
    float vx0 = (x0 >= 0 && x0 < WIN_) ? 1.f : 0.f;
    float vx1 = (x1 >= 0 && x1 < WIN_) ? 1.f : 0.f;
    float w00 = (1.f - wy) * (1.f - wx) * vy0 * vx0;
    float w01 = (1.f - wy) * wx         * vy0 * vx1;
    float w10 = wy * (1.f - wx)         * vy1 * vx0;
    float w11 = wy * wx                 * vy1 * vx1;
    int cy0 = min(max(y0, 0), HIN_ - 1);
    int cy1 = min(max(y1, 0), HIN_ - 1);
    int bx  = min(max(x0, 0), WIN_ - 2);     // both loaded cols in-bounds
    bool sel = (x0 == bx);                   // does col bx hold the x0-corner?
    Param pr;
    pr.w.x = sel ? w00 : w01;   // weight applied to col bx,   row y0
    pr.w.y = sel ? w01 : w00;   // weight applied to col bx+1, row y0
    pr.w.z = sel ? w10 : w11;   // col bx,   row y1
    pr.w.w = sel ? w11 : w10;   // col bx+1, row y1
    pr.o0 = cy0 * WIN_ + bx;
    pr.o1 = cy1 * WIN_ + bx;
    pr.pad0 = 0; pr.pad1 = 0;
    sP[k][p] = pr;
  }
  __syncthreads();

  // producer mapping: pixel pi, 16 channels starting at cg*16
  const int pi = t & 63;
  const int cg = t >> 6;

  // bias folded into accumulator init (bias depends only on cout = lane)
  const int co = cs * 16 + hl;
  const float bv = bias[co];
  v8f acc0, acc1;
  #pragma unroll
  for (int r = 0; r < 8; ++r) { acc0[r] = bv; acc1[r] = 0.f; }

  for (int k = 0; k < KK_; ++k) {
    const int buf = k & 1;
    // ---- stage W_k into ping-pong slot: 8KB = 512 uint4 ----
    {
      const uint4* s4 = (const uint4*)(wbf + k * (COUT_ * CIN_));
      uint4* d4 = (uint4*)sW[buf];
      d4[t]       = s4[t];
      d4[t + 256] = s4[t + 256];
    }
    // ---- gather: 2 x b64 per channel, weights preswapped in sP ----
    {
      Param pr = sP[k][pi];
      const float* xc = xb + cg * 16 * HW_IN;
      union { unsigned short s[8]; uint4 q; } P0, P1;
      #pragma unroll
      for (int j = 0; j < 16; ++j) {
        struct F2 { float a, b; } r0, r1;
        __builtin_memcpy(&r0, xc + pr.o0, sizeof(F2));
        __builtin_memcpy(&r1, xc + pr.o1, sizeof(F2));
        float v = r0.a * pr.w.x + r0.b * pr.w.y + r1.a * pr.w.z + r1.b * pr.w.w;
        bf16_t bvv = (bf16_t)v;
        if (j < 8) P0.s[j] = __builtin_bit_cast(unsigned short, bvv);
        else       P1.s[j - 8] = __builtin_bit_cast(unsigned short, bvv);
        xc += HW_IN;
      }
      uint4* dst = (uint4*)&sA[buf][pi * 64 + cg * 16];   // 32B-aligned
      dst[0] = P0.q;
      dst[1] = P1.q;
    }
    __syncthreads();   // publish buf; prior buffer's readers already drained

    // ---- WMMA: B fragment shared by the wave's two pixel subtiles ----
    #pragma unroll
    for (int kc = 0; kc < 2; ++kc) {
      // B fragment (32x16 KxN): lane -> N=cout=cs*16+hl; K(e) = (e&7)+(e>=8?16:0)+8*hi
      union { v16bf v; uint4 q[2]; } Bm;
      const uint4* b128 =
          (const uint4*)&sW[buf][(cs * 16 + hl) * 64 + kc * 32 + hi * 8];
      Bm.q[0] = b128[0];
      Bm.q[1] = b128[2];
      // A fragments (16x32 MxK): lane -> M=hl; K(e) = (e&7)+(e>=8?16:0)+8*hi
      union { v16bf v; unsigned int u[8]; } A0, A1;
      const unsigned int* a0 =
          (const unsigned int*)&sA[buf][(ps0 * 16 + hl) * 64 + kc * 32 + hi * 8];
      const unsigned int* a1 =
          (const unsigned int*)&sA[buf][(ps1 * 16 + hl) * 64 + kc * 32 + hi * 8];
      #pragma unroll
      for (int vreg = 0; vreg < 4; ++vreg) {
        A0.u[vreg]     = a0[vreg];
        A0.u[vreg + 4] = a0[vreg + 8];
        A1.u[vreg]     = a1[vreg];
        A1.u[vreg + 4] = a1[vreg + 8];
      }
      acc0 = __builtin_amdgcn_wmma_f32_16x16x32_bf16(
          false, A0.v, false, Bm.v, (short)0, acc0, false, false);
      acc1 = __builtin_amdgcn_wmma_f32_16x16x32_bf16(
          false, A1.v, false, Bm.v, (short)0, acc1, false, false);
    }
  }

  // ---- epilogue: D VGPR r -> pixel M=r+8*hi; lane -> cout ----
  #pragma unroll
  for (int s = 0; s < 2; ++s) {
    int ps = s ? ps1 : ps0;
    int p0 = gp0 + ps * 16 + hi * 8;         // 8 consecutive pixels per lane
    int hw2 = p0 & 32767;
    int ho2 = hw2 >> 8;
    int wo2 = hw2 & 255;
    float* op = out + ((b * COUT_ + co) * HOUT_ + ho2) * WOUT_ + wo2;
    float4 q0, q1;
    if (s == 0) {
      q0.x = acc0[0]; q0.y = acc0[1]; q0.z = acc0[2]; q0.w = acc0[3];
      q1.x = acc0[4]; q1.y = acc0[5]; q1.z = acc0[6]; q1.w = acc0[7];
    } else {  // acc1 was initialized to 0; add bias here
      q0.x = acc1[0] + bv; q0.y = acc1[1] + bv; q0.z = acc1[2] + bv; q0.w = acc1[3] + bv;
      q1.x = acc1[4] + bv; q1.y = acc1[5] + bv; q1.z = acc1[6] + bv; q1.w = acc1[7] + bv;
    }
    *(float4*)op       = q0;                 // 16B aligned (wo2 % 8 == 0)
    *(float4*)(op + 4) = q1;
  }
}

extern "C" void kernel_launch(void* const* d_in, const int* in_sizes, int n_in,
                              void* d_out, int out_size, void* d_ws, size_t ws_size,
                              hipStream_t stream) {
  const float* x      = (const float*)d_in[0];
  const float* off    = (const float*)d_in[1];
  const float* weight = (const float*)d_in[2];
  const float* bias   = (const float*)d_in[3];
  float* out = (float*)d_out;
  bf16_t* wbf = (bf16_t*)d_ws;                     // needs 9*64*64*2 = 73,728 B

  dcn_wcvt_kernel<<<(KK_ * COUT_ * CIN_ + 255) / 256, 256, 0, stream>>>(weight, wbf);

  const int npix = B_N * HOUT_ * WOUT_;            // 65536
  dcn_main_kernel<<<npix / PIX_TILE, 256, 0, stream>>>(x, off, wbf, bias, out);
}
